// MogrifierLSTMWrapper_36885179138357
// MI455X (gfx1250) — matrix operand order repaired and checked by reading, not compiled, not run on hardware
//
#include <hip/hip_runtime.h>
#include <hip/hip_bf16.h>
#include <math.h>

// Problem constants (match reference)
#define BQ  64     // batch
#define TT  512    // time steps
#define DIN 256    // layer-0 input dim
#define HH  512    // hidden
#define NL  2      // layers
// Launch config
#define NWG 64
#define TPB 256
#define NWAVES ((NWG * TPB) / 32)   // 512 waves

typedef __attribute__((ext_vector_type(16))) __bf16 v16bf;
typedef __attribute__((ext_vector_type(8)))  __bf16 bf16x8;
typedef __attribute__((ext_vector_type(8)))  float  v8f;

struct KParams {
    const float* x;                // [B, T, DIN] fp32 input
    const float* bias[NL][7];      // fp32 biases (b0..b4, b_ih, b_hh)
    __bf16*      wbf[NL][7];       // bf16 weights [N,K] row-major (W0..W4, W_ih, W_hh)
    __bf16*      ys0;              // [T, B, H] bf16: layer-0 outputs
    float*  xt;  __bf16* xtb;      // [B, D] working x_t (fp32 + bf16 mirror)
    float*  h;   __bf16* hb;       // [B, H] hidden (fp32 + bf16 mirror)
    float*  c;                     // [B, H] cell fp32
    float*  gates;                 // [B, 4H] fp32
    float*  out;                   // [B, T, H] fp32 (d_out)
    float*  hlast;                 // [L, B, H]
    float*  clast;                 // [L, B, H]
    int* barCnt; int* barGen;      // grid barrier state
};

__device__ __forceinline__ float sigf(float v) { return 1.0f / (1.0f + __expf(-v)); }

#define EPI_X0 0   // xt = 2*sig(h W^T + b) * x_layer_input   (also writes xtb)
#define EPI_X  1   // xt = 2*sig(h W^T + b) * xt
#define EPI_H  2   // h  = 2*sig(xt W^T + b) * h              (also writes hb)
#define EPI_G  3   // gates = acc + b_ih + b_hh (stored fp32)

// Device-wide barrier (persistent cooperative kernel, NWG workgroups).
__device__ __forceinline__ void gridSync(int* cnt, int* gen) {
    __syncthreads();
    if (threadIdx.x == 0) {
        __threadfence();
        const int g = __hip_atomic_load(gen, __ATOMIC_RELAXED, __HIP_MEMORY_SCOPE_AGENT);
        const int a = __hip_atomic_fetch_add(cnt, 1, __ATOMIC_ACQ_REL, __HIP_MEMORY_SCOPE_AGENT);
        if (a == NWG - 1) {
            __hip_atomic_store(cnt, 0, __ATOMIC_RELAXED, __HIP_MEMORY_SCOPE_AGENT);
            __hip_atomic_fetch_add(gen, 1, __ATOMIC_RELEASE, __HIP_MEMORY_SCOPE_AGENT);
        } else {
            while (__hip_atomic_load(gen, __ATOMIC_ACQUIRE, __HIP_MEMORY_SCOPE_AGENT) == g) {
                __builtin_amdgcn_s_sleep(2);
            }
        }
        __threadfence();
    }
    __syncthreads();
}

// Stage activation matrix src[64, K] (bf16, row-major, contiguous) into LDS
// with padded row stride KP, starting at column colOff.
__device__ __forceinline__ void stageA(__bf16* lds, int KP, const __bf16* src,
                                       int K, int colOff) {
    const int total = (BQ * K) >> 2;               // 4 bf16 (8B) per chunk
    for (int i = threadIdx.x; i < total; i += TPB) {
        const int e = i << 2;
        const int r = e / K;
        const int cN = e - r * K;
        *(uint2*)(lds + r * KP + colOff + cN) = *(const uint2*)(src + e);
    }
}

// One fused GEMM phase: Y[64,N] = A[64,K] @ W^T (+ optional 2nd operand for gates),
// epilogue applied per 16x16 tile. A comes from LDS, B (weights) straight from L2.
__device__ void gemmPhase(const __bf16* lds, int KP, int K, int N,
                          const __bf16* W, const float* bias,
                          const __bf16* W2, const float* bias2, int K1,
                          int epi, int l, int t, int D,
                          const KParams& p, int gwave) {
    const int lane    = threadIdx.x & 31;
    const int halfSel = lane >> 4;      // 0: lanes 0-15, 1: lanes 16-31
    const int ml      = lane & 15;
    const int numTiles = (BQ / 16) * (N / 16);

    for (int tile = gwave; tile < numTiles; tile += NWAVES) {
        const int mt = tile & 3;        // 64/16 = 4 M-tiles
        const int nt = tile >> 2;
        const int aRow = mt * 16 + ml;  // A: lane holds row m = lane&15
        const int nRow = nt * 16 + ml;  // B: lane holds col n = lane&15 (a W row)
        const __bf16* aBase = lds + aRow * KP + halfSel * 8;
        v8f acc = {0.f, 0.f, 0.f, 0.f, 0.f, 0.f, 0.f, 0.f};

        for (int kk = 0; kk < K; kk += 32) {
            // A fragment: K chunks [kk+hs*8 .. +7] and [kk+16+hs*8 .. +7]
            union { v16bf v; bf16x8 h[2]; } A, B;
            A.h[0] = *(const bf16x8*)(aBase + kk);
            A.h[1] = *(const bf16x8*)(aBase + kk + 16);
            // B fragment: K chunks [kk+hs*8 .. +7] and [kk+16+hs*8 .. +7] of W row nRow
            const int kcol = kk + halfSel * 8;
            const __bf16* wp;
            if (W2 != nullptr && kk >= K1)
                wp = W2 + (size_t)nRow * (K - K1) + (kcol - K1);
            else if (W2 != nullptr)
                wp = W + (size_t)nRow * K1 + kcol;
            else
                wp = W + (size_t)nRow * K + kcol;
            B.h[0] = *(const bf16x8*)(wp);
            B.h[1] = *(const bf16x8*)(wp + 16);
            __builtin_prefetch((const void*)(wp + 128), 0, 0);  // global_prefetch next K slab

            acc = __builtin_amdgcn_wmma_f32_16x16x32_bf16(
                false, A.v, false, B.v, (short)0, acc, false, false);
        }

        // Epilogue. C/D layout: lane col n = lane&15, row m = halfSel*8 + r.
        const int ncol = nt * 16 + ml;
        float bv = bias[ncol];
        if (bias2) bv += bias2[ncol];
        for (int r = 0; r < 8; ++r) {
            const int brow = mt * 16 + halfSel * 8 + r;   // batch row 0..63
            const float v = acc[r] + bv;
            if (epi == EPI_G) {
                p.gates[brow * (4 * HH) + ncol] = v;
            } else {
                const float g = 2.0f * sigf(v);
                if (epi == EPI_H) {
                    const float nh = g * p.h[brow * HH + ncol];
                    p.h[brow * HH + ncol]  = nh;
                    p.hb[brow * HH + ncol] = (__bf16)nh;
                } else {
                    float xs;
                    if (epi == EPI_X0) {
                        xs = (l == 0)
                             ? p.x[((size_t)brow * TT + t) * DIN + ncol]
                             : (float)p.ys0[((size_t)t * BQ + brow) * HH + ncol];
                    } else {
                        xs = p.xt[brow * D + ncol];
                    }
                    const float nx = g * xs;
                    p.xt[brow * D + ncol]  = nx;
                    p.xtb[brow * D + ncol] = (__bf16)nx;
                }
            }
        }
    }
}

__device__ void cellPhase(const KParams& p, int l, int t) {
    for (int i = blockIdx.x * TPB + threadIdx.x; i < BQ * HH; i += NWG * TPB) {
        const int b = i >> 9;        // / 512
        const int j = i & 511;
        const float* g4 = p.gates + b * 4 * HH;
        const float ig = sigf(g4[j]);
        const float fg = sigf(g4[HH + j]);
        const float gg = tanhf(g4[2 * HH + j]);
        const float og = sigf(g4[3 * HH + j]);
        const float c  = fg * p.c[i] + ig * gg;
        const float h  = og * tanhf(c);
        p.c[i] = c; p.h[i] = h; p.hb[i] = (__bf16)h;
        if (l == 0) p.ys0[(size_t)t * BQ * HH + i] = (__bf16)h;
        else        p.out[((size_t)b * TT + t) * HH + j] = h;
        if (t == TT - 1) {
            p.hlast[(size_t)l * BQ * HH + i] = h;
            p.clast[(size_t)l * BQ * HH + i] = c;
        }
    }
}

__global__ void __launch_bounds__(TPB)
cvt_f32_bf16(const float* src, __bf16* dst, int n) {
    const int i = blockIdx.x * blockDim.x + threadIdx.x;
    if (i < n) dst[i] = (__bf16)src[i];
}

__global__ void __launch_bounds__(TPB)
mogrifier_persistent(KParams p) {
    // Max staged A: [xtb | hb] for the gate GEMM, K = 1024 (+8 pad) = 130 KB of the
    // WGP's 320 KB LDS (2 workgroups / WGP still fit).
    __shared__ __bf16 ldsA[BQ * 1040];
    const int wave  = threadIdx.x >> 5;
    const int gwave = blockIdx.x * (TPB / 32) + wave;

    for (int l = 0; l < NL; ++l) {
        const int D = (l == 0) ? DIN : HH;
        // h = c = 0 at layer start
        for (int i = blockIdx.x * TPB + threadIdx.x; i < BQ * HH; i += NWG * TPB) {
            p.h[i] = 0.f; p.c[i] = 0.f; p.hb[i] = (__bf16)0.f;
        }
        gridSync(p.barCnt, p.barGen);

        for (int t = 0; t < TT; ++t) {
            // mog step 0: xt = 2*sig(h W0^T + b0) * x_t      (A = hb, N = D, K = H)
            stageA(ldsA, HH + 8, p.hb, HH, 0); __syncthreads();
            gemmPhase(ldsA, HH + 8, HH, D, p.wbf[l][0], p.bias[l][0],
                      nullptr, nullptr, 0, EPI_X0, l, t, D, p, gwave);
            gridSync(p.barCnt, p.barGen);
            // mog step 1: h = 2*sig(xt W1^T + b1) * h        (A = xtb, N = H, K = D)
            stageA(ldsA, D + 8, p.xtb, D, 0); __syncthreads();
            gemmPhase(ldsA, D + 8, D, HH, p.wbf[l][1], p.bias[l][1],
                      nullptr, nullptr, 0, EPI_H, l, t, D, p, gwave);
            gridSync(p.barCnt, p.barGen);
            // mog step 2
            stageA(ldsA, HH + 8, p.hb, HH, 0); __syncthreads();
            gemmPhase(ldsA, HH + 8, HH, D, p.wbf[l][2], p.bias[l][2],
                      nullptr, nullptr, 0, EPI_X, l, t, D, p, gwave);
            gridSync(p.barCnt, p.barGen);
            // mog step 3
            stageA(ldsA, D + 8, p.xtb, D, 0); __syncthreads();
            gemmPhase(ldsA, D + 8, D, HH, p.wbf[l][3], p.bias[l][3],
                      nullptr, nullptr, 0, EPI_H, l, t, D, p, gwave);
            gridSync(p.barCnt, p.barGen);
            // mog step 4
            stageA(ldsA, HH + 8, p.hb, HH, 0); __syncthreads();
            gemmPhase(ldsA, HH + 8, HH, D, p.wbf[l][4], p.bias[l][4],
                      nullptr, nullptr, 0, EPI_X, l, t, D, p, gwave);
            gridSync(p.barCnt, p.barGen);
            // gates = xt W_ih^T + b_ih + h W_hh^T + b_hh   (A = [xtb|hb], K = D+H, N = 4H)
            const int KG = D + HH;
            stageA(ldsA, KG + 8, p.xtb, D, 0);
            stageA(ldsA, KG + 8, p.hb, HH, D);
            __syncthreads();
            gemmPhase(ldsA, KG + 8, KG, 4 * HH, p.wbf[l][5], p.bias[l][5],
                      p.wbf[l][6], p.bias[l][6], D, EPI_G, l, t, D, p, gwave);
            gridSync(p.barCnt, p.barGen);
            // LSTM cell elementwise
            cellPhase(p, l, t);
            gridSync(p.barCnt, p.barGen);
        }
    }
}

extern "C" void kernel_launch(void* const* d_in, const int* in_sizes, int n_in,
                              void* d_out, int out_size, void* d_ws, size_t ws_size,
                              hipStream_t stream) {
    (void)in_sizes; (void)n_in; (void)out_size; (void)ws_size;
    KParams p{};
    p.x = (const float*)d_in[0];

    char* ws = (char*)d_ws;
    size_t off = 0;
    auto take = [&](size_t bytes) -> char* {
        char* r = ws + off;
        off = (off + bytes + 255) & ~(size_t)255;
        return r;
    };

    // bf16 weight arena
    for (int l = 0; l < NL; ++l) {
        const int din = l ? HH : DIN;
        const int sz[7] = { din * HH, HH * din, din * HH, HH * din, din * HH,
                            4 * HH * din, 4 * HH * HH };
        for (int i = 0; i < 7; ++i) p.wbf[l][i] = (__bf16*)take((size_t)sz[i] * 2);
    }
    p.ys0   = (__bf16*)take((size_t)TT * BQ * HH * 2);
    p.xt    = (float*) take((size_t)BQ * HH * 4);
    p.xtb   = (__bf16*)take((size_t)BQ * HH * 2);
    p.h     = (float*) take((size_t)BQ * HH * 4);
    p.hb    = (__bf16*)take((size_t)BQ * HH * 2);
    p.c     = (float*) take((size_t)BQ * HH * 4);
    p.gates = (float*) take((size_t)BQ * 4 * HH * 4);
    int* bar = (int*)take(256);
    p.barCnt = bar; p.barGen = bar + 1;

    p.out   = (float*)d_out;
    p.hlast = p.out + (size_t)BQ * TT * HH;
    p.clast = p.hlast + (size_t)NL * BQ * HH;

    // d_in layout (setup_inputs order): x, then per layer:
    // W0,b0,W1,b1,W2,b2,W3,b3,W4,b4,W_ih,b_ih,W_hh,b_hh
    for (int l = 0; l < NL; ++l) {
        const int base = 1 + l * 14;
        const int din = l ? HH : DIN;
        const int sz[7] = { din * HH, HH * din, din * HH, HH * din, din * HH,
                            4 * HH * din, 4 * HH * HH };
        const int widx[7] = { base + 0, base + 2, base + 4, base + 6, base + 8,
                              base + 10, base + 12 };
        const int bidx[7] = { base + 1, base + 3, base + 5, base + 7, base + 9,
                              base + 11, base + 13 };
        for (int i = 0; i < 7; ++i) {
            p.bias[l][i] = (const float*)d_in[bidx[i]];
            const int n = sz[i];
            hipLaunchKernelGGL(cvt_f32_bf16, dim3((n + TPB - 1) / TPB), dim3(TPB),
                               0, stream, (const float*)d_in[widx[i]],
                               p.wbf[l][i], n);
        }
    }

    hipMemsetAsync(bar, 0, 256, stream);
    hipLaunchKernelGGL(mogrifier_persistent, dim3(NWG), dim3(TPB), 0, stream, p);
}
